// HypergraphConvolution_46737834115089
// MI455X (gfx1250) — matrix operand order repaired and checked by reading, not compiled, not run on hardware
//
#include <hip/hip_runtime.h>

// ---------------------------------------------------------------------------
// HypergraphConvolution on MI455X (gfx1250, wave32, WMMA)
//   out = (dv ⊙ (H · (de ⊙ (Hᵀ · X)))) · Wᵀ + b
// Three f16->f32 WMMA GEMMs with fused row-scale / bias epilogues.
// Contiguous tile staging uses GLOBAL_LOAD_ASYNC_TO_LDS_B128 (ASYNCcnt path);
// transposing staging keeps the VGPR->LDS route (async copies cannot reorder
// bytes).
// ---------------------------------------------------------------------------

typedef __attribute__((ext_vector_type(16))) _Float16 v16h;
typedef __attribute__((ext_vector_type(8)))  _Float16 v8h;
typedef __attribute__((ext_vector_type(8)))  float    v8f;
typedef __attribute__((ext_vector_type(4)))  int      v4i;

#define BM 128
#define BN 128
#define BK 32
#define LDK 40   // BK + 8 halves pad -> 80B row stride, keeps 16B alignment

#if defined(__has_builtin)
#  if __has_builtin(__builtin_amdgcn_global_load_async_to_lds_b128) && \
      __has_builtin(__builtin_amdgcn_s_wait_asynccnt)
#    define ASYNC_LDS 1
#  endif
#endif
#ifndef ASYNC_LDS
#  define ASYNC_LDS 0
#endif

// ----------------------------- helper kernels ------------------------------

__global__ __launch_bounds__(256) void cvt_f32_f16(const float* __restrict__ s,
                                                   _Float16* __restrict__ d, int n) {
  int i = (blockIdx.x * 256 + threadIdx.x) * 4;
  int stride = gridDim.x * 256 * 4;
  for (; i < n; i += stride) {
    float4 v = *(const float4*)(s + i);
    d[i + 0] = (_Float16)v.x;
    d[i + 1] = (_Float16)v.y;
    d[i + 2] = (_Float16)v.z;
    d[i + 3] = (_Float16)v.w;
  }
}

// dv[row] = rsqrt(sum_e H[row, e])
__global__ __launch_bounds__(256) void row_degree_rsqrt(const float* __restrict__ H,
                                                        float* __restrict__ dv, int E) {
  __shared__ float red[256];
  const float* p = H + (size_t)blockIdx.x * E;
  float s = 0.f;
  for (int e = threadIdx.x; e < E; e += 256) s += p[e];
  red[threadIdx.x] = s;
  __syncthreads();
  for (int off = 128; off > 0; off >>= 1) {
    if ((int)threadIdx.x < off) red[threadIdx.x] += red[threadIdx.x + off];
    __syncthreads();
  }
  if (threadIdx.x == 0) dv[blockIdx.x] = rsqrtf(red[0]);
}

// de[e] = 1 / sum_n H[n, e]
__global__ __launch_bounds__(256) void col_degree_inv(const float* __restrict__ H,
                                                      float* __restrict__ de,
                                                      int Nrows, int E) {
  int e = blockIdx.x * 256 + threadIdx.x;
  if (e >= E) return;
  float s = 0.f;
  for (int n = 0; n < Nrows; ++n) s += H[(size_t)n * E + e];
  de[e] = 1.0f / s;
}

// ------------------------------- WMMA GEMM ---------------------------------
// C[m,n] = scale?(rowScale[m]) * sum_k A(m,k)*B(k,n)  (+ bias[n])
//   AT=0: A(m,k)=A[m*lda+k]      AT=1: A(m,k)=A[k*lda+m]
//   BT=0: B(k,n)=B[k*ldb+n]      BT=1: B(k,n)=B[n*ldb+k]
// Block: 256 threads = 8 waves. Block tile 128x128, K-step 32.
// Wave (wid&1, wid>>1) owns 64x32 -> 4x2 tiles of v_wmma_f32_16x16x32_f16.
template <bool AT, bool BT, bool OUT16, bool SCALE, bool BIAS>
__global__ __launch_bounds__(256) void gemm_wmma_f16(
    const _Float16* __restrict__ A, const _Float16* __restrict__ B,
    void* __restrict__ Cout, const float* __restrict__ rowScale,
    const float* __restrict__ bias, int M, int Nn, int K,
    int lda, int ldb, int ldc) {
  __shared__ _Float16 As[BM * LDK];  // As[m][k]
  __shared__ _Float16 Bs[BN * LDK];  // Bs[n][k]

  const int tid  = threadIdx.x;
  const int lane = tid & 31;
  const int wid  = tid >> 5;
  const int wm   = (wid & 1) * 64;   // wave row base within block tile
  const int wn   = (wid >> 1) * 32;  // wave col base within block tile
  const int m0   = blockIdx.x * BM;
  const int n0   = blockIdx.y * BN;
  const int lrow = lane & 15;
  const int lhi  = lane >> 4;

  v8f acc[4][2];
#pragma unroll
  for (int mt = 0; mt < 4; ++mt)
#pragma unroll
    for (int nt = 0; nt < 2; ++nt)
#pragma unroll
      for (int r = 0; r < 8; ++r) acc[mt][nt][r] = 0.f;

  for (int kt = 0; kt < K; kt += BK) {
    // ---- stage A tile -> As[m][k] ----
    if (!AT) {
#pragma unroll
      for (int it = 0; it < 2; ++it) {
        int c  = tid + it * 256;     // 512 chunks of 8 halves
        int m  = c >> 2;             // 0..127
        int kc = (c & 3) * 8;        // 0,8,16,24
        const _Float16* gp = A + (size_t)(m0 + m) * lda + kt + kc;
#if ASYNC_LDS
        __builtin_amdgcn_global_load_async_to_lds_b128(
            (v4i*)gp, (v4i*)&As[m * LDK + kc], 0, 0);
#else
        *(v8h*)(&As[m * LDK + kc]) = *(const v8h*)gp;
        if (kt + BK < K) __builtin_prefetch((const void*)(gp + BK), 0, 1);
#endif
      }
    } else {
#pragma unroll
      for (int it = 0; it < 2; ++it) {
        int c  = tid + it * 256;
        int k  = c >> 4;             // 0..31
        int mc = (c & 15) * 8;       // 0..120
        const _Float16* gp = A + (size_t)(kt + k) * lda + m0 + mc;
        v8h v = *(const v8h*)gp;
        if (kt + BK < K) __builtin_prefetch((const void*)(gp + (size_t)BK * lda), 0, 1);
#pragma unroll
        for (int i = 0; i < 8; ++i) As[(mc + i) * LDK + k] = v[i];
      }
    }
    // ---- stage B tile -> Bs[n][k] ----
    if (BT) {
#pragma unroll
      for (int it = 0; it < 2; ++it) {
        int c  = tid + it * 256;
        int n  = c >> 2;
        int kc = (c & 3) * 8;
        const _Float16* gp = B + (size_t)(n0 + n) * ldb + kt + kc;
#if ASYNC_LDS
        __builtin_amdgcn_global_load_async_to_lds_b128(
            (v4i*)gp, (v4i*)&Bs[n * LDK + kc], 0, 0);
#else
        *(v8h*)(&Bs[n * LDK + kc]) = *(const v8h*)gp;
        if (kt + BK < K) __builtin_prefetch((const void*)(gp + BK), 0, 1);
#endif
      }
    } else {
#pragma unroll
      for (int it = 0; it < 2; ++it) {
        int c  = tid + it * 256;
        int k  = c >> 4;
        int nc = (c & 15) * 8;
        const _Float16* gp = B + (size_t)(kt + k) * ldb + n0 + nc;
        v8h v = *(const v8h*)gp;
        if (kt + BK < K) __builtin_prefetch((const void*)(gp + (size_t)BK * ldb), 0, 1);
#pragma unroll
        for (int i = 0; i < 8; ++i) Bs[(nc + i) * LDK + k] = v[i];
      }
    }
#if ASYNC_LDS
    if (!AT || BT) __builtin_amdgcn_s_wait_asynccnt(0);  // drain async tile copies
#endif
    __syncthreads();

    // ---- fragments (exact wave32 WMMA layouts, 2x b128 LDS reads each) ----
    // A 16x32 f16: lane M=lrow; halves h<8 -> K=lhi*8+h, h>=8 -> K=16+lhi*8+h-8
    v16h af[4];
#pragma unroll
    for (int mt = 0; mt < 4; ++mt) {
      const _Float16* p = &As[(wm + mt * 16 + lrow) * LDK + lhi * 8];
      v8h lo = *(const v8h*)p;
      v8h hi = *(const v8h*)(p + 16);
#pragma unroll
      for (int i = 0; i < 8; ++i) { af[mt][i] = lo[i]; af[mt][i + 8] = hi[i]; }
    }
    // B 32x16 f16: lane N=lrow; halves h<8 -> K=lhi*8+h, h>=8 -> K=16+lhi*8+h-8
    v16h bf[2];
#pragma unroll
    for (int nt = 0; nt < 2; ++nt) {
      const _Float16* p = &Bs[(wn + nt * 16 + lrow) * LDK + lhi * 8];
      v8h lo = *(const v8h*)p;
      v8h hi = *(const v8h*)(p + 16);
#pragma unroll
      for (int i = 0; i < 8; ++i) { bf[nt][i] = lo[i]; bf[nt][i + 8] = hi[i]; }
    }

#pragma unroll
    for (int mt = 0; mt < 4; ++mt)
#pragma unroll
      for (int nt = 0; nt < 2; ++nt)
        acc[mt][nt] = __builtin_amdgcn_wmma_f32_16x16x32_f16(
            false, af[mt], false, bf[nt], (short)0, acc[mt][nt], false, false);
    __syncthreads();
  }

  // ---- epilogue: C/D layout lane l, reg r -> M = r + (l>>4)*8, N = l&15 ----
#pragma unroll
  for (int mt = 0; mt < 4; ++mt) {
#pragma unroll
    for (int nt = 0; nt < 2; ++nt) {
#pragma unroll
      for (int r = 0; r < 8; ++r) {
        int m = m0 + wm + mt * 16 + lhi * 8 + r;
        int n = n0 + wn + nt * 16 + lrow;
        float v = acc[mt][nt][r];
        if (SCALE) v *= rowScale[m];
        if (BIAS)  v += bias[n];
        if (OUT16) ((_Float16*)Cout)[(size_t)m * ldc + n] = (_Float16)v;
        else       ((float*)Cout)[(size_t)m * ldc + n] = v;
      }
    }
  }
}

// ------------------------------- launcher ----------------------------------

extern "C" void kernel_launch(void* const* d_in, const int* in_sizes, int n_in,
                              void* d_out, int out_size, void* d_ws, size_t ws_size,
                              hipStream_t stream) {
  const float* X = (const float*)d_in[0];  // [N, F]
  const float* H = (const float*)d_in[1];  // [N, E]
  const float* W = (const float*)d_in[2];  // [F, F]
  const float* b = (const float*)d_in[3];  // [F]
  float* out = (float*)d_out;              // [N, F]

  const int F = in_sizes[3];               // 256
  const int N = in_sizes[0] / F;           // 8192
  const int E = in_sizes[1] / N;           // 2048

  // workspace carve-up (f16 copies + intermediates + degree vectors)
  char* ws = (char*)d_ws;
  _Float16* Hh = (_Float16*)ws;                       // N*E
  _Float16* Xh = Hh + (size_t)N * E;                  // N*F
  _Float16* Wh = Xh + (size_t)N * F;                  // F*F
  _Float16* Mh = Wh + (size_t)F * F;                  // E*F
  _Float16* Yh = Mh + (size_t)E * F;                  // N*F
  float* dv = (float*)(Yh + (size_t)N * F);           // N
  float* de = dv + N;                                 // E

  // f32 -> f16 conversions (H is exactly {0,1} -> lossless)
  cvt_f32_f16<<<4096, 256, 0, stream>>>(H, Hh, N * E);
  cvt_f32_f16<<<1024, 256, 0, stream>>>(X, Xh, N * F);
  cvt_f32_f16<<<64,   256, 0, stream>>>(W, Wh, F * F);

  // degree scalings
  row_degree_rsqrt<<<N, 256, 0, stream>>>(H, dv, E);
  col_degree_inv<<<(E + 255) / 256, 256, 0, stream>>>(H, de, N, E);

  // G1: Mh[e,f] = de[e] * sum_n Hh[n,e] * Xh[n,f]   (A = Hᵀ -> AT, lda = E)
  gemm_wmma_f16<true, false, true, true, false>
      <<<dim3(E / BM, F / BN), 256, 0, stream>>>(Hh, Xh, Mh, de, nullptr,
                                                 E, F, N, E, F, F);
  // G2: Yh[n,f] = dv[n] * sum_e Hh[n,e] * Mh[e,f]
  gemm_wmma_f16<false, false, true, true, false>
      <<<dim3(N / BM, F / BN), 256, 0, stream>>>(Hh, Mh, Yh, dv, nullptr,
                                                 N, F, E, E, F, F);
  // G3: out[n,f] = sum_k Yh[n,k] * W[f,k] + b[f]    (B = Wᵀ -> BT, ldb = F)
  gemm_wmma_f16<false, true, false, false, true>
      <<<dim3(N / BM, F / BN), 256, 0, stream>>>(Yh, Wh, out, nullptr, b,
                                                 N, F, F, F, F, F);
}
